// ConditionalRoutedAttention_55044300865626
// MI455X (gfx1250) — matrix operand order repaired and checked by reading, not compiled, not run on hardware
//
#include <hip/hip_runtime.h>
#include <cmath>
#include <cstdint>

// ---------------- problem constants ----------------
#define BATCH   4
#define NTOK    8192
#define DMODEL  1024
#define HEADS   8
#define DH      64
#define HID     512     // HEADS*DH
#define WIN     128
#define NQ      1024
#define NKV     2048
#define SCALE   0.125f  // DH^-0.5

typedef __bf16 bf16_t;
typedef __attribute__((ext_vector_type(16))) __bf16 v16bf;
typedef __attribute__((ext_vector_type(8)))  __bf16 v8bf;
typedef __attribute__((ext_vector_type(8)))  float  v8f;

__device__ __forceinline__ v8f vzero() {
    v8f z;
#pragma unroll
    for (int i = 0; i < 8; ++i) z[i] = 0.f;
    return z;
}

// D = A(16x32 bf16) * B(32x16 bf16) + C(16x16 f32)
__device__ __forceinline__ v8f wmma_bf16(v16bf a, v16bf b, v8f c) {
    return __builtin_amdgcn_wmma_f32_16x16x32_bf16(false, a, false, b, (short)0, c, false, false);
}

// A fragment: 16x32 tile, row-major source with leading dim lda (elements).
// Contiguous k, as in frag_b: lane L holds row M=L&15; kbase=(L>>4)*16; elems 0..7 -> K=kbase..kbase+7,
// elems 8..15 -> K=kbase+8..kbase+15  => two contiguous 16B loads.
__device__ __forceinline__ v16bf frag_a(const bf16_t* A, int lda, int lane) {
    const bf16_t* p = A + (size_t)(lane & 15) * lda + ((lane >> 4) << 4);
    v8bf lo = *(const v8bf*)p;
    v8bf hi = *(const v8bf*)(p + 8);
    v16bf f;
#pragma unroll
    for (int i = 0; i < 8; ++i) { f[i] = lo[i]; f[i + 8] = hi[i]; }
    return f;
}

// B fragment: 32x16 tile given as BT[n][k] (K-contiguous rows, leading dim ldb).
// Contiguous k, as in frag_a: lane L holds col N=L&15, K = (L>>4)*16 .. +15 => one contiguous 32B load.
__device__ __forceinline__ v16bf frag_b(const bf16_t* BT, int ldb, int lane) {
    const bf16_t* p = BT + (size_t)(lane & 15) * ldb + ((lane >> 4) << 4);
    return *(const v16bf*)p;
}

__device__ __forceinline__ float wave_sum(float v) {
    v += __shfl_xor(v, 16, 32); v += __shfl_xor(v, 8, 32);
    v += __shfl_xor(v, 4, 32);  v += __shfl_xor(v, 2, 32);
    v += __shfl_xor(v, 1, 32);
    return v;
}

// ---------------- CDNA5 async global->LDS DMA (ASYNCcnt path) ----------------
__device__ __forceinline__ unsigned lds_off(const void* p) {
    // generic pointers to LDS carry the LDS offset in addr[31:0] (ISA 10.2)
    return (unsigned)(uintptr_t)p;
}
__device__ __forceinline__ void async_copy_b128(unsigned lds_addr, const void* gaddr) {
    asm volatile("global_load_async_to_lds_b128 %0, %1, off"
                 :: "v"(lds_addr), "v"(gaddr) : "memory");
}
__device__ __forceinline__ void wait_async() {
    asm volatile("s_wait_asynccnt 0x0" ::: "memory");
}

// ---------------- 1) routing scores: s = x . token ----------------
__global__ __launch_bounds__(256) void route_score_kernel(const float* __restrict__ x,
                                                          const float* __restrict__ tq,
                                                          const float* __restrict__ tkv,
                                                          float* __restrict__ sq,
                                                          float* __restrict__ skv) {
    const int lane = threadIdx.x & 31, wid = threadIdx.x >> 5;
    const size_t row = (size_t)blockIdx.x * 8 + wid;   // < BATCH*NTOK
    const float* xr = x + row * DMODEL;
    float aq = 0.f, akv = 0.f;
#pragma unroll 8
    for (int j = 0; j < DMODEL / 32; ++j) {
        float v = xr[j * 32 + lane];
        aq  += v * tq[j * 32 + lane];
        akv += v * tkv[j * 32 + lane];
    }
    aq = wave_sum(aq); akv = wave_sum(akv);
    if (lane == 0) { sq[row] = aq; skv[row] = akv; }
}

// ---------------- 2) coordinate-descent relaxed top-k (EPS = 1) ----------------
__device__ __forceinline__ float block_reduce(float v, bool is_max, float* red) {
#pragma unroll
    for (int m = 16; m >= 1; m >>= 1) {
        float o = __shfl_xor(v, m, 32);
        v = is_max ? fmaxf(v, o) : v + o;
    }
    __syncthreads();
    if ((threadIdx.x & 31) == 0) red[threadIdx.x >> 5] = v;
    __syncthreads();
    if (threadIdx.x < 32) {
        float r = red[threadIdx.x];
#pragma unroll
        for (int m = 16; m >= 1; m >>= 1) {
            float o = __shfl_xor(r, m, 32);
            r = is_max ? fmaxf(r, o) : r + o;
        }
        if (threadIdx.x == 0) red[0] = r;
    }
    __syncthreads();
    return red[0];
}

__global__ __launch_bounds__(1024) void coor_descent_kernel(const float* __restrict__ sq,
                                                            const float* __restrict__ skv,
                                                            float* __restrict__ scq,
                                                            float* __restrict__ sckv,
                                                            float logkq, float logkkv) {
    __shared__ float red[32];
    const int route = blockIdx.x >> 2, b = blockIdx.x & 3;
    const float* sin = (route ? skv : sq) + (size_t)b * NTOK;
    float* sout = (route ? sckv : scq) + (size_t)b * NTOK;
    const float logk = route ? logkkv : logkq;
    float s[8], bb[8];
#pragma unroll
    for (int j = 0; j < 8; ++j) { s[j] = sin[j * 1024 + threadIdx.x]; bb[j] = -s[j]; }
    float a = 0.f;
    for (int it = 0; it < 50; ++it) {
        float lm = s[0] + bb[0];
#pragma unroll
        for (int j = 1; j < 8; ++j) lm = fmaxf(lm, s[j] + bb[j]);
        float m = block_reduce(lm, true, red);
        float ls = 0.f;
#pragma unroll
        for (int j = 0; j < 8; ++j) ls += __expf(s[j] + bb[j] - m);
        float tot = block_reduce(ls, false, red);
        a = logk - (m + __logf(tot));                 // EPS == 1
#pragma unroll
        for (int j = 0; j < 8; ++j) bb[j] = -fmaxf(s[j] + a, 0.f);
    }
#pragma unroll
    for (int j = 0; j < 8; ++j) sout[j * 1024 + threadIdx.x] = __expf(s[j] + a + bb[j]);
}

// ---------------- 3) stable descending top-k by ranking ----------------
__global__ __launch_bounds__(256) void topk_kernel(const float* __restrict__ scores,
                                                   int* __restrict__ idx_out, int Kcnt) {
    __shared__ __align__(16) float sc[NTOK];          // 32 KB
    const int b = blockIdx.x / (NTOK / 256);
    const int seg = blockIdx.x % (NTOK / 256);
    // async DMA the whole score row into LDS (2048 x 16B chunks, 8 per thread)
    const unsigned sc_base = lds_off(&sc[0]);
    const char* row = (const char*)(scores + (size_t)b * NTOK);
#pragma unroll
    for (int i = 0; i < 8; ++i) {
        int e = threadIdx.x + i * 256;                // 16-byte chunk id
        async_copy_b128(sc_base + (unsigned)e * 16, row + (size_t)e * 16);
    }
    wait_async();
    __syncthreads();
    const int my = seg * 256 + threadIdx.x;
    const float v = sc[my];
    int rank = 0;
    for (int j = 0; j < NTOK; ++j) {
        float o = sc[j];
        rank += (o > v) || (o == v && j < my);        // matches jax top_k tie order
    }
    if (rank < Kcnt) idx_out[(size_t)b * Kcnt + rank] = my;
}

// ---------------- 4) rmsnorm (+gamma) -> bf16 ----------------
__global__ __launch_bounds__(256) void rmsnorm_cast_kernel(const float* __restrict__ x,
                                                           const float* __restrict__ gamma,
                                                           bf16_t* __restrict__ out) {
    const int lane = threadIdx.x & 31, wid = threadIdx.x >> 5;
    const size_t row = (size_t)blockIdx.x * 8 + wid;
    const float* xr = x + row * DMODEL;
    float ss = 0.f;
#pragma unroll 8
    for (int j = 0; j < DMODEL / 32; ++j) { float v = xr[j * 32 + lane]; ss += v * v; }
    ss = wave_sum(ss);
    const float scl = 32.f / fmaxf(sqrtf(ss), 1e-12f); // sqrt(D)=32
#pragma unroll 8
    for (int j = 0; j < DMODEL / 32; ++j) {
        int c = j * 32 + lane;
        out[row * DMODEL + c] = (bf16_t)(xr[c] * scl * gamma[c]);
    }
}

// gather routed rows + rmsnorm(gamma_heavy) -> bf16
__global__ __launch_bounds__(256) void gather_norm_kernel(const float* __restrict__ x,
                                                          const float* __restrict__ gamma,
                                                          const int* __restrict__ idxq,
                                                          const int* __restrict__ idxkv,
                                                          bf16_t* __restrict__ xqn,
                                                          bf16_t* __restrict__ ctxn) {
    const int lane = threadIdx.x & 31, wid = threadIdx.x >> 5;
    const size_t g = (size_t)blockIdx.x * 8 + wid;    // < BATCH*(NQ+NKV)
    const float* src;
    bf16_t* dst;
    if (g < (size_t)BATCH * NQ) {
        int b = (int)(g / NQ);
        src = x + ((size_t)b * NTOK + idxq[g]) * DMODEL;
        dst = xqn + g * DMODEL;
    } else {
        size_t g2 = g - (size_t)BATCH * NQ;
        int b = (int)(g2 / NKV);
        src = x + ((size_t)b * NTOK + idxkv[g2]) * DMODEL;
        dst = ctxn + g2 * DMODEL;
    }
    float ss = 0.f;
#pragma unroll 8
    for (int j = 0; j < DMODEL / 32; ++j) { float v = src[j * 32 + lane]; ss += v * v; }
    ss = wave_sum(ss);
    const float scl = 32.f / fmaxf(sqrtf(ss), 1e-12f);
#pragma unroll 8
    for (int j = 0; j < DMODEL / 32; ++j) {
        int c = j * 32 + lane;
        dst[c] = (bf16_t)(src[c] * scl * gamma[c]);
    }
}

// ---------------- 5) weight prep ----------------
__global__ __launch_bounds__(256) void transpose_cast_kernel(const float* __restrict__ src,
                                                             bf16_t* __restrict__ dst, int R, int C) {
    int i = blockIdx.x * 256 + threadIdx.x;
    if (i >= R * C) return;
    int r = i / C, c = i % C;
    dst[(size_t)c * R + r] = (bf16_t)src[i];
}
__global__ __launch_bounds__(256) void cast_kernel(const float* __restrict__ src,
                                                   bf16_t* __restrict__ dst, int n) {
    int i = blockIdx.x * 256 + threadIdx.x;
    if (i < n) dst[i] = (bf16_t)src[i];
}

// ---------------- 6) WMMA GEMM: C[M,N] = A[M,K] * BT[N,K]^T ----------------
// block = 256 threads (8 waves, 4x2), block tile 128x128, wave tile 32x64.
template <int OUT_BF16>
__global__ __launch_bounds__(256) void gemm_kernel(const bf16_t* __restrict__ A,
                                                   const bf16_t* __restrict__ BT,
                                                   void* __restrict__ Cout,
                                                   int M, int Nn, int K) {
    const int lane = threadIdx.x & 31, wid = threadIdx.x >> 5;
    const int wm = wid & 3, wn = wid >> 2;
    const int m0 = blockIdx.y * 128 + wm * 32;
    const int n0 = blockIdx.x * 128 + wn * 64;
    const bf16_t* Abase = A + (size_t)m0 * K;
    const bf16_t* Bbase = BT + (size_t)n0 * K;
    v8f acc[2][4];
#pragma unroll
    for (int i = 0; i < 2; ++i)
#pragma unroll
        for (int t = 0; t < 4; ++t) acc[i][t] = vzero();

    for (int k0 = 0; k0 < K; k0 += 32) {
        if (k0 + 64 < K) {
            __builtin_prefetch(Abase + k0 + 64, 0, 0);
            __builtin_prefetch(Bbase + k0 + 64, 0, 0);
        }
        v16bf a0 = frag_a(Abase + k0, K, lane);
        v16bf a1 = frag_a(Abase + (size_t)16 * K + k0, K, lane);
#pragma unroll
        for (int t = 0; t < 4; ++t) {
            v16bf bfrag = frag_b(Bbase + (size_t)t * 16 * K + k0, K, lane);
            acc[0][t] = wmma_bf16(a0, bfrag, acc[0][t]);
            acc[1][t] = wmma_bf16(a1, bfrag, acc[1][t]);
        }
    }
    const int nc = lane & 15, mb = (lane >> 4) << 3;
#pragma unroll
    for (int i = 0; i < 2; ++i)
#pragma unroll
        for (int t = 0; t < 4; ++t)
#pragma unroll
            for (int r = 0; r < 8; ++r) {
                size_t row = (size_t)m0 + i * 16 + mb + r;
                size_t col = (size_t)n0 + t * 16 + nc;
                if (OUT_BF16) ((bf16_t*)Cout)[row * Nn + col] = (bf16_t)acc[i][t][r];
                else          ((float*)Cout)[row * Nn + col] = acc[i][t][r];
            }
}

// ---------------- 7) light local attention (one block per b,window,head) ----------------
// S kept in registers; softmax row-reduced with shfl_xor inside the lane-half
// that owns each C-layout row; V chunk DMA'd to LDS (async) overlapped with S WMMAs.
__global__ __launch_bounds__(256) void local_attn_kernel(const bf16_t* __restrict__ qL,
                                                         const bf16_t* __restrict__ kvL,
                                                         bf16_t* __restrict__ attnL) {
    const int lane = threadIdx.x & 31, wid = threadIdx.x >> 5;
    const int h = blockIdx.x & 7, w = (blockIdx.x >> 3) & 63, b = blockIdx.x >> 9;
    const bf16_t* Qb = qL  + ((size_t)b * NTOK + (size_t)w * WIN) * HID       + h * DH;
    const bf16_t* Kb = kvL + ((size_t)b * NTOK + (size_t)w * WIN) * (2 * HID) + h * DH;
    const bf16_t* Vb = Kb + HID;

    __shared__ __align__(32) bf16_t Pb[WIN][136];   // P in A-loadable layout
    __shared__ __align__(32) bf16_t Vraw[WIN][DH];  // async DMA target (row-major V)
    __shared__ __align__(32) bf16_t Vt[DH][144];    // V^T so B-frags are contiguous

    // async DMA: V window (128 x 64 bf16) -> LDS, overlapped with S computation
    const unsigned vraw_base = lds_off(&Vraw[0][0]);
#pragma unroll
    for (int i = 0; i < 4; ++i) {
        int e = threadIdx.x + i * 256;              // 16-byte chunk id (1024 total)
        int k = e >> 3, c8 = e & 7;
        async_copy_b128(vraw_base + (unsigned)e * 16,
                        (const char*)(Vb + (size_t)k * (2 * HID)) + c8 * 16);
    }

    const int M0 = wid * 16, nc = lane & 15, mb = (lane >> 4) << 3;
    v16bf aq0 = frag_a(Qb + (size_t)M0 * HID, HID, lane);
    v16bf aq1 = frag_a(Qb + (size_t)M0 * HID + 32, HID, lane);

    v8f S[8];
#pragma unroll
    for (int t = 0; t < 8; ++t) {
        v8f acc = vzero();
        v16bf b0 = frag_b(Kb + (size_t)t * 16 * (2 * HID), 2 * HID, lane);
        v16bf b1 = frag_b(Kb + (size_t)t * 16 * (2 * HID) + 32, 2 * HID, lane);
        acc = wmma_bf16(aq0, b0, acc);
        acc = wmma_bf16(aq1, b1, acc);
        S[t] = acc;
    }
    float rsum[8];
#pragma unroll
    for (int r = 0; r < 8; ++r) {
        float m = S[0][r] * SCALE;
#pragma unroll
        for (int t = 1; t < 8; ++t) m = fmaxf(m, S[t][r] * SCALE);
        m = fmaxf(m, __shfl_xor(m, 1, 32)); m = fmaxf(m, __shfl_xor(m, 2, 32));
        m = fmaxf(m, __shfl_xor(m, 4, 32)); m = fmaxf(m, __shfl_xor(m, 8, 32));
        float s = 0.f;
#pragma unroll
        for (int t = 0; t < 8; ++t) { float p = __expf(S[t][r] * SCALE - m); S[t][r] = p; s += p; }
        s += __shfl_xor(s, 1, 32); s += __shfl_xor(s, 2, 32);
        s += __shfl_xor(s, 4, 32); s += __shfl_xor(s, 8, 32);
        rsum[r] = s;
    }
    wait_async();
    __syncthreads();                                // Vraw visible to all waves
    // LDS transpose Vraw -> Vt, publish P
    for (int e = threadIdx.x; e < WIN * DH; e += 256) {
        int k = e >> 6, d = e & 63;
        Vt[d][k] = Vraw[k][d];
    }
#pragma unroll
    for (int t = 0; t < 8; ++t)
#pragma unroll
        for (int r = 0; r < 8; ++r) Pb[M0 + mb + r][t * 16 + nc] = (bf16_t)S[t][r];
    __syncthreads();

    v8f o[4];
#pragma unroll
    for (int t = 0; t < 4; ++t) o[t] = vzero();
#pragma unroll
    for (int k0 = 0; k0 < WIN; k0 += 32) {
        v16bf ap = frag_a(&Pb[0][0] + (size_t)M0 * 136 + k0, 136, lane);
#pragma unroll
        for (int t = 0; t < 4; ++t) {
            v16bf bv = frag_b(&Vt[0][0] + (size_t)t * 16 * 144 + k0, 144, lane);
            o[t] = wmma_bf16(ap, bv, o[t]);
        }
    }
    bf16_t* Ob = attnL + ((size_t)b * NTOK + (size_t)w * WIN) * HID + h * DH;
#pragma unroll
    for (int t = 0; t < 4; ++t)
#pragma unroll
        for (int r = 0; r < 8; ++r)
            Ob[(size_t)(M0 + mb + r) * HID + t * 16 + nc] = (bf16_t)(o[t][r] / rsum[r]);
}

// ---------------- 8) heavy routed attention, flash-style over NKV ----------------
__global__ __launch_bounds__(256) void heavy_attn_kernel(const bf16_t* __restrict__ qH,
                                                         const bf16_t* __restrict__ kvH,
                                                         bf16_t* __restrict__ attnH) {
    const int lane = threadIdx.x & 31, wid = threadIdx.x >> 5;
    const int qt = blockIdx.x & 7, h = (blockIdx.x >> 3) & 7, b = blockIdx.x >> 6;
    const bf16_t* Qb = qH + ((size_t)b * NQ + (size_t)qt * 128) * HID + h * DH;

    __shared__ __align__(32) bf16_t Pb[128][136];
    __shared__ __align__(32) bf16_t Vraw[128][DH];
    __shared__ __align__(32) bf16_t Vt[DH][144];

    const int M0 = wid * 16, nc = lane & 15, mb = (lane >> 4) << 3;
    const unsigned vraw_base = lds_off(&Vraw[0][0]);
    v16bf aq0 = frag_a(Qb + (size_t)M0 * HID, HID, lane);
    v16bf aq1 = frag_a(Qb + (size_t)M0 * HID + 32, HID, lane);

    v8f o[4];
#pragma unroll
    for (int t = 0; t < 4; ++t) o[t] = vzero();
    float rmax[8], rsum[8];
#pragma unroll
    for (int r = 0; r < 8; ++r) { rmax[r] = -1e30f; rsum[r] = 0.f; }

    for (int c = 0; c < NKV / 128; ++c) {
        __syncthreads();                            // Pb/Vt/Vraw reuse barrier
        const bf16_t* Kb = kvH + ((size_t)b * NKV + (size_t)c * 128) * (2 * HID) + h * (2 * DH);
        const bf16_t* Vb = Kb + DH;
        // async DMA this chunk's V -> LDS; overlaps with the S WMMAs below
#pragma unroll
        for (int i = 0; i < 4; ++i) {
            int e = threadIdx.x + i * 256;
            int k = e >> 3, c8 = e & 7;
            async_copy_b128(vraw_base + (unsigned)e * 16,
                            (const char*)(Vb + (size_t)k * (2 * HID)) + c8 * 16);
        }
        v8f S[8];
#pragma unroll
        for (int t = 0; t < 8; ++t) {
            v8f acc = vzero();
            v16bf b0 = frag_b(Kb + (size_t)t * 16 * (2 * HID), 2 * HID, lane);
            v16bf b1 = frag_b(Kb + (size_t)t * 16 * (2 * HID) + 32, 2 * HID, lane);
            acc = wmma_bf16(aq0, b0, acc);
            acc = wmma_bf16(aq1, b1, acc);
            S[t] = acc;
        }
        float fr[8];
#pragma unroll
        for (int r = 0; r < 8; ++r) {
            float m = S[0][r] * SCALE;
#pragma unroll
            for (int t = 1; t < 8; ++t) m = fmaxf(m, S[t][r] * SCALE);
            m = fmaxf(m, __shfl_xor(m, 1, 32)); m = fmaxf(m, __shfl_xor(m, 2, 32));
            m = fmaxf(m, __shfl_xor(m, 4, 32)); m = fmaxf(m, __shfl_xor(m, 8, 32));
            float nm = fmaxf(rmax[r], m);
            float f = __expf(rmax[r] - nm);
            float s = 0.f;
#pragma unroll
            for (int t = 0; t < 8; ++t) { float p = __expf(S[t][r] * SCALE - nm); S[t][r] = p; s += p; }
            s += __shfl_xor(s, 1, 32); s += __shfl_xor(s, 2, 32);
            s += __shfl_xor(s, 4, 32); s += __shfl_xor(s, 8, 32);
            rsum[r] = rsum[r] * f + s;
            rmax[r] = nm;
            fr[r] = f;
        }
        wait_async();
        __syncthreads();                            // Vraw complete + visible
        for (int e = threadIdx.x; e < 128 * DH; e += 256) {
            int k = e >> 6, d = e & 63;
            Vt[d][k] = Vraw[k][d];
        }
#pragma unroll
        for (int t = 0; t < 8; ++t)
#pragma unroll
            for (int r = 0; r < 8; ++r) Pb[M0 + mb + r][t * 16 + nc] = (bf16_t)S[t][r];
        __syncthreads();                            // Vt + Pb ready
#pragma unroll
        for (int t = 0; t < 4; ++t)
#pragma unroll
            for (int r = 0; r < 8; ++r) o[t][r] *= fr[r];      // online-softmax rescale
#pragma unroll
        for (int k0 = 0; k0 < 128; k0 += 32) {
            v16bf ap = frag_a(&Pb[0][0] + (size_t)M0 * 136 + k0, 136, lane);
#pragma unroll
            for (int t = 0; t < 4; ++t) {
                v16bf bv = frag_b(&Vt[0][0] + (size_t)t * 16 * 144 + k0, 144, lane);
                o[t] = wmma_bf16(ap, bv, o[t]);
            }
        }
    }
    bf16_t* Ob = attnH + ((size_t)b * NQ + (size_t)qt * 128) * HID + h * DH;
#pragma unroll
    for (int t = 0; t < 4; ++t)
#pragma unroll
        for (int r = 0; r < 8; ++r)
            Ob[(size_t)(M0 + mb + r) * HID + t * 16 + nc] = (bf16_t)(o[t][r] / rsum[r]);
}

// ---------------- 10) scatter heavy rows into output ----------------
__global__ __launch_bounds__(256) void scatter_add_kernel(const float* __restrict__ oH,
                                                          const int* __restrict__ idxq,
                                                          float* __restrict__ out) {
    const int g = blockIdx.x;                 // < BATCH*NQ
    const int b = g / NQ;
    const int dstrow = idxq[g];
    float* o = out + ((size_t)b * NTOK + dstrow) * DMODEL;
    const float* src = oH + (size_t)g * DMODEL;
    for (int c = threadIdx.x; c < DMODEL; c += 256) o[c] += src[c];
}

// ---------------- launcher ----------------
extern "C" void kernel_launch(void* const* d_in, const int* in_sizes, int n_in,
                              void* d_out, int out_size, void* d_ws, size_t ws_size,
                              hipStream_t stream) {
    (void)in_sizes; (void)n_in; (void)out_size; (void)ws_size;
    const float* x      = (const float*)d_in[0];
    const float* gammaL = (const float*)d_in[1];
    const float* wqL    = (const float*)d_in[2];
    const float* wkvL   = (const float*)d_in[3];
    const float* woL    = (const float*)d_in[4];
    const float* tq     = (const float*)d_in[5];
    const float* tkv    = (const float*)d_in[6];
    const float* gammaH = (const float*)d_in[7];
    const float* wqH    = (const float*)d_in[8];
    const float* wkvH   = (const float*)d_in[9];   // [2*HID, D] torch layout == BT layout
    const float* woH    = (const float*)d_in[10];
    float* out = (float*)d_out;

    char* wsp = (char*)d_ws;
    auto take = [&](size_t bytes) -> char* {
        char* r = wsp;
        wsp += (bytes + 255) & ~((size_t)255);
        return r;
    };
    float* sq     = (float*)take((size_t)BATCH * NTOK * 4);
    float* skv    = (float*)take((size_t)BATCH * NTOK * 4);
    float* scq    = (float*)take((size_t)BATCH * NTOK * 4);
    float* sckv   = (float*)take((size_t)BATCH * NTOK * 4);
    int*   idxq   = (int*)take((size_t)BATCH * NQ * 4);
    int*   idxkv  = (int*)take((size_t)BATCH * NKV * 4);
    bf16_t* xnL    = (bf16_t*)take((size_t)BATCH * NTOK * DMODEL * 2);
    bf16_t* qLb    = (bf16_t*)take((size_t)BATCH * NTOK * HID * 2);
    bf16_t* kvLb   = (bf16_t*)take((size_t)BATCH * NTOK * 2 * HID * 2);
    bf16_t* attnLb = (bf16_t*)take((size_t)BATCH * NTOK * HID * 2);
    bf16_t* wqLT   = (bf16_t*)take((size_t)DMODEL * HID * 2);
    bf16_t* wkvLT  = (bf16_t*)take((size_t)DMODEL * 2 * HID * 2);
    bf16_t* woLT   = (bf16_t*)take((size_t)HID * DMODEL * 2);
    bf16_t* wqHT   = (bf16_t*)take((size_t)DMODEL * HID * 2);
    bf16_t* wkvHb  = (bf16_t*)take((size_t)2 * HID * DMODEL * 2);
    bf16_t* woHT   = (bf16_t*)take((size_t)HID * DMODEL * 2);
    bf16_t* xqn    = (bf16_t*)take((size_t)BATCH * NQ * DMODEL * 2);
    bf16_t* ctxn   = (bf16_t*)take((size_t)BATCH * NKV * DMODEL * 2);
    bf16_t* qHb    = (bf16_t*)take((size_t)BATCH * NQ * HID * 2);
    bf16_t* kvHb   = (bf16_t*)take((size_t)BATCH * NKV * 2 * HID * 2);
    bf16_t* attnHb = (bf16_t*)take((size_t)BATCH * NQ * HID * 2);
    float*  oH     = (float*)take((size_t)BATCH * NQ * DMODEL * 4);

    const int BN = BATCH * NTOK;

    // routing
    route_score_kernel<<<BN / 8, 256, 0, stream>>>(x, tq, tkv, sq, skv);
    coor_descent_kernel<<<8, 1024, 0, stream>>>(sq, skv, scq, sckv,
                                                logf(1152.f), logf(2304.f)); // k = n_tokens*9/8
    topk_kernel<<<BATCH * (NTOK / 256), 256, 0, stream>>>(scq, idxq, NQ);
    topk_kernel<<<BATCH * (NTOK / 256), 256, 0, stream>>>(sckv, idxkv, NKV);

    // weight prep (bf16, K-contiguous BT layouts)
    transpose_cast_kernel<<<(DMODEL * HID + 255) / 256, 256, 0, stream>>>(wqL, wqLT, DMODEL, HID);
    transpose_cast_kernel<<<(DMODEL * 2 * HID + 255) / 256, 256, 0, stream>>>(wkvL, wkvLT, DMODEL, 2 * HID);
    transpose_cast_kernel<<<(HID * DMODEL + 255) / 256, 256, 0, stream>>>(woL, woLT, HID, DMODEL);
    transpose_cast_kernel<<<(DMODEL * HID + 255) / 256, 256, 0, stream>>>(wqH, wqHT, DMODEL, HID);
    transpose_cast_kernel<<<(HID * DMODEL + 255) / 256, 256, 0, stream>>>(woH, woHT, HID, DMODEL);
    cast_kernel<<<(2 * HID * DMODEL + 255) / 256, 256, 0, stream>>>(wkvH, wkvHb, 2 * HID * DMODEL);

    // light path
    rmsnorm_cast_kernel<<<BN / 8, 256, 0, stream>>>(x, gammaL, xnL);
    gemm_kernel<1><<<dim3(HID / 128, BN / 128), 256, 0, stream>>>(xnL, wqLT, qLb, BN, HID, DMODEL);
    gemm_kernel<1><<<dim3(2 * HID / 128, BN / 128), 256, 0, stream>>>(xnL, wkvLT, kvLb, BN, 2 * HID, DMODEL);
    local_attn_kernel<<<BATCH * (NTOK / WIN) * HEADS, 256, 0, stream>>>(qLb, kvLb, attnLb);
    gemm_kernel<0><<<dim3(DMODEL / 128, BN / 128), 256, 0, stream>>>(attnLb, woLT, out, BN, DMODEL, HID);

    // heavy path (routed scores are exactly 1.0 forward via straight-through)
    gather_norm_kernel<<<(BATCH * (NQ + NKV)) / 8, 256, 0, stream>>>(x, gammaH, idxq, idxkv, xqn, ctxn);
    gemm_kernel<1><<<dim3(HID / 128, (BATCH * NQ) / 128), 256, 0, stream>>>(xqn, wqHT, qHb, BATCH * NQ, HID, DMODEL);
    gemm_kernel<1><<<dim3(2 * HID / 128, (BATCH * NKV) / 128), 256, 0, stream>>>(ctxn, wkvHb, kvHb, BATCH * NKV, 2 * HID, DMODEL);
    heavy_attn_kernel<<<BATCH * HEADS * (NQ / 128), 256, 0, stream>>>(qHb, kvHb, attnHb);
    gemm_kernel<0><<<dim3(DMODEL / 128, (BATCH * NQ) / 128), 256, 0, stream>>>(attnHb, woHT, oH, BATCH * NQ, DMODEL, HID);
    scatter_add_kernel<<<BATCH * NQ, 256, 0, stream>>>(oH, idxq, out);
}
